// CombinedNeckHeadQueryGenWrapper_84499186581925
// MI455X (gfx1250) — matrix operand order repaired and checked by reading, not compiled, not run on hardware
//
#include <hip/hip_runtime.h>
#include <hip/hip_bf16.h>
#include <math.h>
#include <stdint.h>

// ---------------------------------------------------------------------------
// CDNA5 / gfx1250 CenterPoint-style neck+head+query pipeline.
// Matrix math: v_wmma_f32_16x16x32_bf16, wave32, 16x32 C tile per wave
// (2 accumulators reuse one A fragment). Weight panels are staged into LDS
// with global_load_async_to_lds_b128 (ASYNCcnt), double-buffered so the DMA
// of tap t+1 overlaps the WMMA of tap t; fragments read back as ds_load_b128.
// ---------------------------------------------------------------------------

typedef __bf16 bf16_t;
typedef __bf16 v8bf  __attribute__((ext_vector_type(8)));
typedef __bf16 v16bf __attribute__((ext_vector_type(16)));
typedef float  v8f   __attribute__((ext_vector_type(8)));
typedef int    v4i   __attribute__((ext_vector_type(4)));

typedef __attribute__((address_space(1))) v4i gas_v4i;   // global 128-bit chunk
typedef __attribute__((address_space(3))) v4i las_v4i;   // LDS 128-bit chunk

#define HH 180
#define HW_N 32400          // 180*180
#define HW2_N 8100          // 90*90
#define C_IN 256
#define C0 128
#define C1 256
#define C_NECK 512
#define C_HD 64
#define NTASK 6
#define POST_MAX 500
#define FINAL_K 300

__constant__ int c_nc[NTASK] = {1, 2, 2, 1, 2, 2};

// ------------------------------- WMMA helpers ------------------------------
__device__ __forceinline__ v16bf bf_zero16() {
  v16bf r;
#pragma unroll
  for (int i = 0; i < 16; ++i) r[i] = (bf16_t)0.0f;
  return r;
}

// A-matrix fragment (16x32 bf16): lanes 0-15 hold row M=lane, K {0..7,16..23};
// lanes 16-31 hold row M=lane-16, K {8..15,24..31}.  p = row base + k0.
__device__ __forceinline__ v16bf load_a_frag(const bf16_t* p, int laneHi) {
  const v8bf lo = *reinterpret_cast<const v8bf*>(p + (laneHi ? 8 : 0));
  const v8bf hi = *reinterpret_cast<const v8bf*>(p + (laneHi ? 8 : 0) + 16);
  v16bf r;
#pragma unroll
  for (int i = 0; i < 8; ++i) { r[i] = lo[i]; r[i + 8] = hi[i]; }
  return r;
}

// B-matrix fragment (32x16 bf16): lane n holds column N=n&15,
// lanes 0-15 -> K {0..7,16..23}, lanes 16-31 -> K {8..15,24..31}.
__device__ __forceinline__ v16bf load_b_frag(const bf16_t* p, int laneHi) {
  const v8bf lo = *reinterpret_cast<const v8bf*>(p + (laneHi ? 8 : 0));
  const v8bf hi = *reinterpret_cast<const v8bf*>(p + (laneHi ? 8 : 0) + 16);
  v16bf r;
#pragma unroll
  for (int i = 0; i < 8; ++i) { r[i] = lo[i]; r[i + 8] = hi[i]; }
  return r;
}

__device__ __forceinline__ v8f wmma_bf16(v16bf a, v16bf b, v8f c) {
  return __builtin_amdgcn_wmma_f32_16x16x32_bf16(false, a, false, b, (short)0, c,
                                                 false, false);
}

__device__ __forceinline__ v8f f_zero8() {
  v8f r;
#pragma unroll
  for (int i = 0; i < 8; ++i) r[i] = 0.0f;
  return r;
}

// ---- async global->LDS copy (16 bytes / issue), tracked by ASYNCcnt --------
__device__ __forceinline__ void cp_async16(bf16_t* ldst, const bf16_t* gsrc) {
#if __has_builtin(__builtin_amdgcn_global_load_async_to_lds_b128)
  __builtin_amdgcn_global_load_async_to_lds_b128(
      (gas_v4i*)(uintptr_t)gsrc,
      (las_v4i*)(unsigned)(uintptr_t)ldst,
      0, 0);
#else
  asm volatile("global_load_async_to_lds_b128 %0, %1, off"
               :
               : "v"((unsigned)(uintptr_t)ldst), "v"(gsrc)
               : "memory");
#endif
}

__device__ __forceinline__ void cp_async_wait() {
#if __has_builtin(__builtin_amdgcn_s_wait_asynccnt)
  __builtin_amdgcn_s_wait_asynccnt(0);
#else
  asm volatile("s_wait_asynccnt 0" ::: "memory");
#endif
}

__device__ __forceinline__ float sigmoidf_(float x) { return 1.0f / (1.0f + expf(-x)); }

// --------------------------- layout / prep kernels -------------------------
// NCHW f32 -> [HW, C] bf16
__global__ void k_nchw_to_hwc_bf16(const float* __restrict__ in, bf16_t* __restrict__ out,
                                   int C, int HW) {
  long idx = (long)blockIdx.x * blockDim.x + threadIdx.x;
  if (idx >= (long)C * HW) return;
  int p = (int)(idx / C), c = (int)(idx % C);
  out[idx] = (bf16_t)in[(long)c * HW + p];
}

// [O][I][KH][KW] f32 -> [KH*KW][O][I] bf16 (K-contiguous B matrix per tap)
__global__ void k_prep_conv_w(const float* __restrict__ w, bf16_t* __restrict__ out,
                              int O, int I, int KH, int KW) {
  long idx = (long)blockIdx.x * blockDim.x + threadIdx.x;
  long total = (long)O * I * KH * KW;
  if (idx >= total) return;
  int kw = (int)(idx % KW); long t = idx / KW;
  int kh = (int)(t % KH); t /= KH;
  int i = (int)(t % I); int o = (int)(t / I);
  out[(((long)(kh * KW + kw) * O + o) * I) + i] = (bf16_t)w[idx];
}

__global__ void k_f32_to_bf16(const float* __restrict__ in, bf16_t* __restrict__ out, long n) {
  long idx = (long)blockIdx.x * blockDim.x + threadIdx.x;
  if (idx < n) out[idx] = (bf16_t)in[idx];
}

// Pack per-task 1x1 heads into a [32 x 64] bf16 B matrix (rows 12..31 zero)
// rows: 0-1 hm, 2-3 reg, 4 hei, 5-7 dim, 8-9 rot, 10-11 vel.
__global__ void k_prep_head(const float* __restrict__ w_hm, const float* __restrict__ w_reg,
                            const float* __restrict__ w_hei, const float* __restrict__ w_dim,
                            const float* __restrict__ w_rot, const float* __restrict__ w_vel,
                            const float* __restrict__ b_hm, const float* __restrict__ b_reg,
                            const float* __restrict__ b_hei, const float* __restrict__ b_dim,
                            const float* __restrict__ b_rot, const float* __restrict__ b_vel,
                            bf16_t* __restrict__ Wh, float* __restrict__ Bh) {
  int t = blockIdx.x;
  int tid = threadIdx.x;                 // 1024 threads
  int r = tid >> 6, i = tid & 63;        // r in [0,16)
  float v = 0.0f;
  if (r < 2)        v = w_hm [((long)t * 2 + r)      * 64 + i];
  else if (r < 4)   v = w_reg[((long)t * 2 + r - 2)  * 64 + i];
  else if (r < 5)   v = w_hei[(long)t * 64 + i];
  else if (r < 8)   v = w_dim[((long)t * 3 + r - 5)  * 64 + i];
  else if (r < 10)  v = w_rot[((long)t * 2 + r - 8)  * 64 + i];
  else if (r < 12)  v = w_vel[((long)t * 2 + r - 10) * 64 + i];
  Wh[((long)t * 32 + r) * 64 + i] = (bf16_t)v;
  Wh[((long)t * 32 + r + 16) * 64 + i] = (bf16_t)0.0f;   // pad rows 16..31
  if (i == 0) {
    float b = 0.0f;
    if (r < 2)       b = b_hm [t * 2 + r];
    else if (r < 4)  b = b_reg[t * 2 + r - 2];
    else if (r < 5)  b = b_hei[t];
    else if (r < 8)  b = b_dim[t * 3 + r - 5];
    else if (r < 10) b = b_rot[t * 2 + r - 8];
    else if (r < 12) b = b_vel[t * 2 + r - 10];
    Bh[t * 16 + r] = b;
  }
}

// ----------------------- WMMA implicit-GEMM convolution ---------------------
// A: [Hin*Win, Cin] bf16 (channels-last).  Wt: [ks*ks][Cout][Cin] bf16.
// Block = 4 waves; wave w owns M-tile (blockIdx.x*4+w), block owns 32 N cols.
// Double-buffered LDS: tap t+1's 32xCin panel is async-DMA'd while tap t's
// WMMAs run.  Dynamic LDS = 2 * 32*Cin*2 bytes.
__global__ void __launch_bounds__(128)
k_conv_wmma(const bf16_t* __restrict__ A, int Hin, int Win, int Cin,
            const bf16_t* __restrict__ Wt, const float* __restrict__ bias,
            float* __restrict__ Cf, bf16_t* __restrict__ Cb,
            int Hout, int Wout, int Cout, int ldc, int coff,
            int ksize, int stride, int pad, int relu) {
  extern __shared__ bf16_t sB[];                 // [2][32*Cin]
  const int panel = 32 * Cin;
  const int lane = threadIdx.x & 31;
  const int wave = threadIdx.x >> 5;
  const int M = Hout * Wout;
  const int mt = blockIdx.x * 4 + wave;
  const bool active = (mt * 16 < M);
  const int n0 = blockIdx.y * 32;
  const int lane15 = lane & 15, laneHi = lane >> 4;
  const int mrow = mt * 16 + lane15;
  const int oy = mrow / Wout, ox = mrow % Wout;
  v8f acc0 = f_zero8(), acc1 = f_zero8();
  const int taps = ksize * ksize;
  const int chunks = panel >> 3;                 // 8 bf16 per 16B async issue
  // prologue: stage tap 0 into buffer 0
  {
    const bf16_t* src = Wt + ((long)0 * Cout + n0) * Cin;
    for (int i = threadIdx.x; i < chunks; i += 128)
      cp_async16(sB + (long)i * 8, src + (long)i * 8);
  }
  for (int t = 0; t < taps; ++t) {
    const int cur = t & 1;
    cp_async_wait();                              // panel[cur] landed
    __syncthreads();
    if (t + 1 < taps) {                           // overlap next tap's DMA
      const bf16_t* src = Wt + ((long)(t + 1) * Cout + n0) * Cin;
      bf16_t* dst = sB + (long)(1 - cur) * panel;
      for (int i = threadIdx.x; i < chunks; i += 128)
        cp_async16(dst + (long)i * 8, src + (long)i * 8);
    }
    // ---- WMMA over K from panel[cur] ----
    const int ty = t / ksize, tx = t % ksize;
    const int iy = oy * stride + ty - pad;
    const int ix = ox * stride + tx - pad;
    const bool inb = active && (mrow < M) && (iy >= 0) && (iy < Hin) &&
                     (ix >= 0) && (ix < Win);
    const bf16_t* arow = inb ? (A + ((long)iy * Win + ix) * Cin) : nullptr;
    const bf16_t* b0 = sB + (long)cur * panel + (long)lane15 * Cin;
    const bf16_t* b1 = sB + (long)cur * panel + (long)(16 + lane15) * Cin;
    for (int k0 = 0; k0 < Cin; k0 += 32) {
      if (inb) __builtin_prefetch(arow + k0 + 64, 0, 1);
      v16bf a = bf_zero16();
      if (inb) a = load_a_frag(arow + k0, laneHi);
      v16bf f0 = load_b_frag(b0 + k0, laneHi);   // ds_load from LDS
      v16bf f1 = load_b_frag(b1 + k0, laneHi);
      acc0 = wmma_bf16(a, f0, acc0);
      acc1 = wmma_bf16(a, f1, acc1);
    }
  }
  if (!active) return;
  const int na = n0 + lane15, nb = n0 + 16 + lane15;
  const float bia = bias ? bias[na] : 0.0f;
  const float bib = bias ? bias[nb] : 0.0f;
#pragma unroll
  for (int r = 0; r < 8; ++r) {
    int m = mt * 16 + r + 8 * laneHi;            // C layout: VGPR r, laneHi half
    if (m < M) {
      float va = acc0[r] + bia;
      float vb = acc1[r] + bib;
      if (relu) { va = fmaxf(va, 0.0f); vb = fmaxf(vb, 0.0f); }
      long oa = (long)m * ldc + coff + na;
      long ob = (long)m * ldc + coff + nb;
      if (Cf) { Cf[oa] = va; Cf[ob] = vb; }
      if (Cb) { Cb[oa] = (bf16_t)va; Cb[ob] = (bf16_t)vb; }
    }
  }
}

// Transposed-conv 2x2 stride2: 4 parity GEMMs with scattered stores into neck.
__global__ void __launch_bounds__(128)
k_deconv_wmma(const bf16_t* __restrict__ A /*[8100,256]*/,
              const bf16_t* __restrict__ Wd /*[4][256][256]*/,
              const float* __restrict__ bias,
              float* __restrict__ neck_f, bf16_t* __restrict__ neck_b) {
  __shared__ bf16_t sB[32 * C1];
  const int lane = threadIdx.x & 31;
  const int wave = threadIdx.x >> 5;
  const int M = HW2_N;
  const int mt = blockIdx.x * 4 + wave;
  const bool active = (mt * 16 < M);
  const int n0 = blockIdx.y * 32;
  const int ab = blockIdx.z, pa = ab >> 1, pb = ab & 1;
  const int lane15 = lane & 15, laneHi = lane >> 4;
  const int mrow = mt * 16 + lane15;
  // stage B panel
  {
    const bf16_t* src = Wd + ((long)ab * C1 + n0) * C1;
    const int chunks = (32 * C1) >> 3;
    for (int i = threadIdx.x; i < chunks; i += 128)
      cp_async16(sB + (long)i * 8, src + (long)i * 8);
    cp_async_wait();
  }
  __syncthreads();
  const bf16_t* arow = A + (long)mrow * C1;
  const bf16_t* b0 = sB + (long)lane15 * C1;
  const bf16_t* b1 = sB + (long)(16 + lane15) * C1;
  v8f acc0 = f_zero8(), acc1 = f_zero8();
  for (int k0 = 0; k0 < C1; k0 += 32) {
    v16bf a = bf_zero16();
    if (active && mrow < M) a = load_a_frag(arow + k0, laneHi);
    v16bf f0 = load_b_frag(b0 + k0, laneHi);
    v16bf f1 = load_b_frag(b1 + k0, laneHi);
    acc0 = wmma_bf16(a, f0, acc0);
    acc1 = wmma_bf16(a, f1, acc1);
  }
  if (!active) return;
  const int na = n0 + lane15, nb = n0 + 16 + lane15;
  const float bia = bias[na], bib = bias[nb];
#pragma unroll
  for (int r = 0; r < 8; ++r) {
    int m = mt * 16 + r + 8 * laneHi;
    if (m < M) {
      int y = m / 90, x = m % 90;
      long base = ((long)(2 * y + pa) * HH + (2 * x + pb)) * C_NECK + 256;
      float va = fmaxf(acc0[r] + bia, 0.0f);
      float vb = fmaxf(acc1[r] + bib, 0.0f);
      neck_f[base + na] = va; neck_b[base + na] = (bf16_t)va;
      neck_f[base + nb] = vb; neck_b[base + nb] = (bf16_t)vb;
    }
  }
}

// Generic GEMM: C[M,Ncols] = A[M,K] * B[Npad,K]^T (+bias)(+residual)(+=accum)
// Block owns 32 B-rows (padded); stores guarded by Ncols.  Dynamic LDS = 64*K B.
__global__ void __launch_bounds__(128)
k_gemm_wmma(const bf16_t* __restrict__ A, int M, int K, int lda,
            const bf16_t* __restrict__ B, int ldb,
            const float* __restrict__ bias, const float* __restrict__ Res, int ldres,
            float* __restrict__ Cf, int ldc, int Ncols, int accum, int relu) {
  extern __shared__ bf16_t sB[];                 // [32*K]
  const int lane = threadIdx.x & 31;
  const int wave = threadIdx.x >> 5;
  const int mt = blockIdx.x * 4 + wave;
  const bool active = (mt * 16 < M);
  const int n0 = blockIdx.y * 32;
  const int lane15 = lane & 15, laneHi = lane >> 4;
  const int mrow = mt * 16 + lane15;
  // stage B panel (rows contiguous since ldb == K for all callers)
  {
    const bf16_t* src = B + (long)n0 * ldb;
    const int chunks = (32 * K) >> 3;
    for (int i = threadIdx.x; i < chunks; i += 128)
      cp_async16(sB + (long)i * 8, src + (long)i * 8);
    cp_async_wait();
  }
  __syncthreads();
  const bf16_t* arow = A + (long)mrow * lda;
  const bf16_t* b0 = sB + (long)lane15 * K;
  const bf16_t* b1 = sB + (long)(16 + lane15) * K;
  v8f acc0 = f_zero8(), acc1 = f_zero8();
  for (int k0 = 0; k0 < K; k0 += 32) {
    if (active) __builtin_prefetch(arow + k0 + 64, 0, 1);
    v16bf a = bf_zero16();
    if (active && mrow < M) a = load_a_frag(arow + k0, laneHi);
    v16bf f0 = load_b_frag(b0 + k0, laneHi);
    v16bf f1 = load_b_frag(b1 + k0, laneHi);
    acc0 = wmma_bf16(a, f0, acc0);
    acc1 = wmma_bf16(a, f1, acc1);
  }
  if (!active) return;
  const int na = n0 + lane15, nb = n0 + 16 + lane15;
#pragma unroll
  for (int r = 0; r < 8; ++r) {
    int m = mt * 16 + r + 8 * laneHi;
    if (m < M) {
      if (na < Ncols) {
        float v = acc0[r] + (bias ? bias[na] : 0.0f);
        if (Res) v += Res[(long)m * ldres + na];
        if (relu) v = fmaxf(v, 0.0f);
        long o = (long)m * ldc + na;
        if (accum) Cf[o] += v; else Cf[o] = v;
      }
      if (nb < Ncols) {
        float v = acc1[r] + (bias ? bias[nb] : 0.0f);
        if (Res) v += Res[(long)m * ldres + nb];
        if (relu) v = fmaxf(v, 0.0f);
        long o = (long)m * ldc + nb;
        if (accum) Cf[o] += v; else Cf[o] = v;
      }
    }
  }
}

// ------------------------------- decode path -------------------------------
__global__ void k_scores(const float* __restrict__ head_f, float* __restrict__ scores) {
  int p = blockIdx.x * blockDim.x + threadIdx.x;
  if (p >= HW_N) return;
  int t = blockIdx.z, c = blockIdx.y;
  float out = 0.0f;
  if (c < c_nc[t]) {
    const float* hf = head_f + (long)t * HW_N * 16;
    int y = p / HH, x = p % HH;
    float ctr = sigmoidf_(hf[(long)p * 16 + c]);
    float mx = ctr;
    for (int dy = -1; dy <= 1; ++dy)
      for (int dx = -1; dx <= 1; ++dx) {
        int yy = y + dy, xx = x + dx;
        if (yy >= 0 && yy < HH && xx >= 0 && xx < HH) {
          float v = sigmoidf_(hf[((long)yy * HH + xx) * 16 + c]);
          mx = fmaxf(mx, v);
        }
      }
    out = (ctr == mx) ? ctr : 0.0f;
  }
  scores[((long)t * 2 + c) * HW_N + p] = out;
}

__global__ void k_topk_task(float* __restrict__ scores, float* __restrict__ top_s,
                            int* __restrict__ top_i) {
  __shared__ float sv[256];
  __shared__ int si[256];
  const int t = blockIdx.x;
  const int n = c_nc[t] * HW_N;
  float* sc = scores + (long)t * 2 * HW_N;
  for (int k = 0; k < POST_MAX; ++k) {
    float best = -1e30f; int bi = 0;
    for (int i = threadIdx.x; i < n; i += 256) {
      float v = sc[i];
      if (v > best) { best = v; bi = i; }
    }
    sv[threadIdx.x] = best; si[threadIdx.x] = bi;
    __syncthreads();
    for (int s = 128; s > 0; s >>= 1) {
      if (threadIdx.x < s && sv[threadIdx.x + s] > sv[threadIdx.x]) {
        sv[threadIdx.x] = sv[threadIdx.x + s];
        si[threadIdx.x] = si[threadIdx.x + s];
      }
      __syncthreads();
    }
    if (threadIdx.x == 0) {
      top_s[t * POST_MAX + k] = sv[0];
      top_i[t * POST_MAX + k] = si[0];
      sc[si[0]] = -1e30f;
    }
    __syncthreads();
  }
}

__global__ void k_topk_final(const float* __restrict__ top_s, float* __restrict__ fin_s,
                             int* __restrict__ fin_j) {
  __shared__ float fv[NTASK * POST_MAX];
  __shared__ float sv[256];
  __shared__ int si[256];
  for (int i = threadIdx.x; i < NTASK * POST_MAX; i += 256) fv[i] = top_s[i];
  __syncthreads();
  for (int k = 0; k < FINAL_K; ++k) {
    float best = -1e30f; int bi = 0;
    for (int i = threadIdx.x; i < NTASK * POST_MAX; i += 256) {
      if (fv[i] > best) { best = fv[i]; bi = i; }
    }
    sv[threadIdx.x] = best; si[threadIdx.x] = bi;
    __syncthreads();
    for (int s = 128; s > 0; s >>= 1) {
      if (threadIdx.x < s && sv[threadIdx.x + s] > sv[threadIdx.x]) {
        sv[threadIdx.x] = sv[threadIdx.x + s];
        si[threadIdx.x] = si[threadIdx.x + s];
      }
      __syncthreads();
    }
    if (threadIdx.x == 0) {
      fin_s[k] = sv[0];
      fin_j[k] = si[0];
      fv[si[0]] = -1e30f;
    }
    __syncthreads();
  }
}

__global__ void k_decode_final(const float* __restrict__ head_f, const int* __restrict__ top_i,
                               const float* __restrict__ fin_s, const int* __restrict__ fin_j,
                               float* __restrict__ out_xyz, float* __restrict__ qpred) {
  int q = threadIdx.x + blockIdx.x * blockDim.x;
  if (q >= FINAL_K) return;
  int j = fin_j[q];
  int t = j / POST_MAX;
  int flat = top_i[j];
  int s = flat % HW_N;
  const float* hf = head_f + ((long)t * HW_N + s) * 16;
  float reg0 = hf[2], reg1 = hf[3], hei = hf[4];
  float d0 = expf(hf[5]), d1 = expf(hf[6]), d2 = expf(hf[7]);
  float yaw = atan2f(hf[8], hf[9]);
  float xs = (float)(s % HH), ys = (float)(s / HH);
  float X = (xs + reg0) * 0.6f - 54.0f;   // * OSF*VOXEL + PC
  float Y = (ys + reg1) * 0.6f - 54.0f;
  out_xyz[q * 3 + 0] = X;
  out_xyz[q * 3 + 1] = Y;
  out_xyz[q * 3 + 2] = hei;
  qpred[q * 7 + 0] = d0; qpred[q * 7 + 1] = d1; qpred[q * 7 + 2] = d2;
  qpred[q * 7 + 3] = yaw; qpred[q * 7 + 4] = hf[10]; qpred[q * 7 + 5] = hf[11];
  qpred[q * 7 + 6] = fin_s[q];
}

__global__ void k_gridsample(const float* __restrict__ neck_f, const float* __restrict__ xyz,
                             float* __restrict__ qf, bf16_t* __restrict__ qfb) {
  int q = blockIdx.x;
  int c = blockIdx.y * 256 + threadIdx.x;
  float X = xyz[q * 3 + 0], Y = xyz[q * 3 + 1];
  float xn = (X + 54.0f) / 108.0f * 2.0f - 1.0f;
  float yn = (Y + 54.0f) / 108.0f * 2.0f - 1.0f;
  float gx = (xn + 1.0f) * 90.0f - 0.5f;
  float gy = (yn + 1.0f) * 90.0f - 0.5f;
  float x0f = floorf(gx), y0f = floorf(gy);
  float wx = gx - x0f, wy = gy - y0f;
  int x0 = (int)x0f, y0 = (int)y0f;
  float acc = 0.0f;
#pragma unroll
  for (int tap = 0; tap < 4; ++tap) {
    int xi = x0 + (tap & 1), yi = y0 + (tap >> 1);
    float w = ((tap & 1) ? wx : 1.0f - wx) * ((tap >> 1) ? wy : 1.0f - wy);
    bool valid = (xi >= 0) && (xi < HH) && (yi >= 0) && (yi < HH);
    int xc = min(max(xi, 0), HH - 1), yc = min(max(yi, 0), HH - 1);
    float v = neck_f[((long)yc * HH + xc) * C_NECK + c];
    acc += valid ? v * w : 0.0f;
  }
  qf[(long)q * C_NECK + c] = acc;
  qfb[(long)q * C_NECK + c] = (bf16_t)acc;
}

__global__ void k_pos2embed(const float* __restrict__ qpred, bf16_t* __restrict__ pe) {
  int q = blockIdx.x, f = threadIdx.x;      // 224 threads
  int j = f / 32, i = f % 32;
  float v = qpred[q * 7 + j] * 6.283185307179586f;
  float d = 1.0f + (float)(i >> 1) / 16.0f; // dim_t = 2*(k//2)/32 + 1
  float e = (i & 1) ? cosf(v / d) : sinf(v / d);
  pe[(long)q * 224 + f] = (bf16_t)e;
}

// ------------------------------- host driver -------------------------------
extern "C" void kernel_launch(void* const* d_in, const int* in_sizes, int n_in,
                              void* d_out, int out_size, void* d_ws, size_t ws_size,
                              hipStream_t stream) {
  const float* bev_feat = (const float*)d_in[0];
  const float* w_block0 = (const float*)d_in[1];
  const float* b_block0 = (const float*)d_in[2];
  const float* w_block1 = (const float*)d_in[3];
  const float* b_block1 = (const float*)d_in[4];
  const float* w_neck0  = (const float*)d_in[5];
  const float* b_neck0  = (const float*)d_in[6];
  const float* w_neck1  = (const float*)d_in[7];
  const float* b_neck1  = (const float*)d_in[8];
  const float* w_shared = (const float*)d_in[9];
  const float* b_shared = (const float*)d_in[10];
  const float* w_hm  = (const float*)d_in[11];
  const float* b_hm  = (const float*)d_in[12];
  const float* w_reg = (const float*)d_in[13];
  const float* b_reg = (const float*)d_in[14];
  const float* w_hei = (const float*)d_in[15];
  const float* b_hei = (const float*)d_in[16];
  const float* w_dim = (const float*)d_in[17];
  const float* b_dim = (const float*)d_in[18];
  const float* w_rot = (const float*)d_in[19];
  const float* b_rot = (const float*)d_in[20];
  const float* w_vel = (const float*)d_in[21];
  const float* b_vel = (const float*)d_in[22];
  const float* w_pre = (const float*)d_in[23];
  const float* b_pre = (const float*)d_in[24];
  const float* w_qe  = (const float*)d_in[25];
  const float* b_qe  = (const float*)d_in[26];
  const float* w_qpe = (const float*)d_in[27];
  const float* b_qpe = (const float*)d_in[28];

  float* out_bev = (float*)d_out;                                 // [32400,512]
  float* out_qf  = out_bev + (long)HW_N * C_NECK;                 // [300,512]
  float* out_xyz = out_qf + (long)FINAL_K * C_NECK;               // [300,3]

  // ---- workspace carve-up (256B aligned) ----
  char* ws = (char*)d_ws;
  size_t off = 0;
  auto carve = [&](size_t bytes) -> char* {
    off = (off + 255) & ~(size_t)255;
    char* p = ws + off;
    off += bytes;
    return p;
  };
  bf16_t* bev_bf  = (bf16_t*)carve((size_t)HW_N * C_IN * 2);
  bf16_t* wb0     = (bf16_t*)carve((size_t)9 * C0 * C_IN * 2);
  bf16_t* x0_bf   = (bf16_t*)carve((size_t)HW_N * C0 * 2);
  bf16_t* wb1     = (bf16_t*)carve((size_t)9 * C1 * C0 * 2);
  bf16_t* x1_bf   = (bf16_t*)carve((size_t)HW2_N * C1 * 2);
  bf16_t* wn0     = (bf16_t*)carve((size_t)1 * 256 * C0 * 2);
  bf16_t* wdec    = (bf16_t*)carve((size_t)4 * 256 * 256 * 2);
  float*  neck_f  = (float*)carve((size_t)HW_N * C_NECK * 4);
  bf16_t* neck_bf = (bf16_t*)carve((size_t)HW_N * C_NECK * 2);
  bf16_t* wsh     = (bf16_t*)carve((size_t)NTASK * 9 * C_HD * C_NECK * 2);
  bf16_t* sh_bf   = (bf16_t*)carve((size_t)HW_N * C_HD * 2);
  bf16_t* whead   = (bf16_t*)carve((size_t)NTASK * 32 * C_HD * 2);
  float*  bhead   = (float*)carve((size_t)NTASK * 16 * 4);
  float*  head_f  = (float*)carve((size_t)NTASK * HW_N * 16 * 4);
  float*  scores  = (float*)carve((size_t)NTASK * 2 * HW_N * 4);
  float*  top_s   = (float*)carve((size_t)NTASK * POST_MAX * 4);
  int*    top_i   = (int*)carve((size_t)NTASK * POST_MAX * 4);
  float*  fin_s   = (float*)carve((size_t)FINAL_K * 4);
  int*    fin_j   = (int*)carve((size_t)FINAL_K * 4);
  float*  qpred   = (float*)carve((size_t)FINAL_K * 7 * 4);
  float*  qf_f    = (float*)carve((size_t)FINAL_K * C_NECK * 4);
  bf16_t* qf_bf   = (bf16_t*)carve((size_t)FINAL_K * C_NECK * 2);
  bf16_t* pe_bf   = (bf16_t*)carve((size_t)FINAL_K * 224 * 2);
  bf16_t* wpre_bf = (bf16_t*)carve((size_t)C_NECK * C_NECK * 2);
  bf16_t* wqe_bf  = (bf16_t*)carve((size_t)C_NECK * C_NECK * 2);
  bf16_t* wqpe_bf = (bf16_t*)carve((size_t)C_NECK * 224 * 2);
  (void)ws_size; (void)n_in; (void)in_sizes; (void)out_size;

  auto cdiv = [](long a, long b) { return (int)((a + b - 1) / b); };
  const dim3 B128(128);
  auto conv_shmem = [](int cin) { return (size_t)2 * 32 * cin * 2; };
  auto gemm_shmem = [](int k)   { return (size_t)32 * k * 2; };

  // ---- weight / layout prep ----
  k_nchw_to_hwc_bf16<<<cdiv((long)HW_N * C_IN, 256), 256, 0, stream>>>(bev_feat, bev_bf, C_IN, HW_N);
  k_prep_conv_w<<<cdiv((long)C0 * C_IN * 9, 256), 256, 0, stream>>>(w_block0, wb0, C0, C_IN, 3, 3);
  k_prep_conv_w<<<cdiv((long)C1 * C0 * 9, 256), 256, 0, stream>>>(w_block1, wb1, C1, C0, 3, 3);
  k_prep_conv_w<<<cdiv((long)256 * C0, 256), 256, 0, stream>>>(w_neck0, wn0, 256, C0, 1, 1);
  k_prep_conv_w<<<cdiv((long)256 * 256 * 4, 256), 256, 0, stream>>>(w_neck1, wdec, 256, 256, 2, 2);
  for (int t = 0; t < NTASK; ++t) {
    k_prep_conv_w<<<cdiv((long)C_HD * C_NECK * 9, 256), 256, 0, stream>>>(
        w_shared + (long)t * C_HD * C_NECK * 9, wsh + (long)t * 9 * C_HD * C_NECK,
        C_HD, C_NECK, 3, 3);
  }
  k_prep_head<<<NTASK, 1024, 0, stream>>>(w_hm, w_reg, w_hei, w_dim, w_rot, w_vel,
                                          b_hm, b_reg, b_hei, b_dim, b_rot, b_vel,
                                          whead, bhead);
  k_f32_to_bf16<<<cdiv((long)C_NECK * C_NECK, 256), 256, 0, stream>>>(w_pre, wpre_bf, (long)C_NECK * C_NECK);
  k_f32_to_bf16<<<cdiv((long)C_NECK * C_NECK, 256), 256, 0, stream>>>(w_qe, wqe_bf, (long)C_NECK * C_NECK);
  k_f32_to_bf16<<<cdiv((long)C_NECK * 224, 256), 256, 0, stream>>>(w_qpe, wqpe_bf, (long)C_NECK * 224);

  // ---- backbone / neck (WMMA) ----
  const int mt180 = (HW_N + 15) / 16;   // 2025
  const int mt90  = (HW2_N + 15) / 16;  // 507
  // block0: 256->128, 3x3 s1
  k_conv_wmma<<<dim3(cdiv(mt180, 4), C0 / 32), B128, conv_shmem(C_IN), stream>>>(
      bev_bf, HH, HH, C_IN, wb0, b_block0, nullptr, x0_bf, HH, HH, C0, C0, 0, 3, 1, 1, 1);
  // block1: 128->256, 3x3 s2
  k_conv_wmma<<<dim3(cdiv(mt90, 4), C1 / 32), B128, conv_shmem(C0), stream>>>(
      x0_bf, HH, HH, C0, wb1, b_block1, nullptr, x1_bf, 90, 90, C1, C1, 0, 3, 2, 1, 1);
  // neck0: 1x1 128->256 into neck[:,0:256]
  k_conv_wmma<<<dim3(cdiv(mt180, 4), 256 / 32), B128, conv_shmem(C0), stream>>>(
      x0_bf, HH, HH, C0, wn0, b_neck0, neck_f, neck_bf, HH, HH, 256, C_NECK, 0, 1, 1, 0, 1);
  // neck1: deconv 2x2 s2 into neck[:,256:512]
  k_deconv_wmma<<<dim3(cdiv(mt90, 4), 256 / 32, 4), B128, 0, stream>>>(
      x1_bf, wdec, b_neck1, neck_f, neck_bf);

  // ---- per-task shared conv + heads (WMMA) ----
  for (int t = 0; t < NTASK; ++t) {
    k_conv_wmma<<<dim3(cdiv(mt180, 4), C_HD / 32), B128, conv_shmem(C_NECK), stream>>>(
        neck_bf, HH, HH, C_NECK, wsh + (long)t * 9 * C_HD * C_NECK, b_shared + t * C_HD,
        nullptr, sh_bf, HH, HH, C_HD, C_HD, 0, 3, 1, 1, 1);
    k_gemm_wmma<<<dim3(cdiv(mt180, 4), 1), B128, gemm_shmem(C_HD), stream>>>(
        sh_bf, HW_N, C_HD, C_HD, whead + (long)t * 32 * C_HD, C_HD,
        bhead + t * 16, nullptr, 0, head_f + (long)t * HW_N * 16, 16, 16, 0, 0);
  }

  // ---- decode ----
  k_scores<<<dim3(cdiv(HW_N, 256), 2, NTASK), 256, 0, stream>>>(head_f, scores);
  k_topk_task<<<NTASK, 256, 0, stream>>>(scores, top_s, top_i);
  k_topk_final<<<1, 256, 0, stream>>>(top_s, fin_s, fin_j);
  k_decode_final<<<1, 320, 0, stream>>>(head_f, top_i, fin_s, fin_j, out_xyz, qpred);
  k_gridsample<<<dim3(FINAL_K, 2), 256, 0, stream>>>(neck_f, out_xyz, qf_f, qf_bf);
  k_pos2embed<<<FINAL_K, 224, 0, stream>>>(qpred, pe_bf);

  // ---- output GEMMs (WMMA) ----
  // bev_out = neck @ w_pre^T + b_pre + neck
  k_gemm_wmma<<<dim3(cdiv(mt180, 4), C_NECK / 32), B128, gemm_shmem(C_NECK), stream>>>(
      neck_bf, HW_N, C_NECK, C_NECK, wpre_bf, C_NECK, b_pre, neck_f, C_NECK,
      out_bev, C_NECK, C_NECK, 0, 0);
  // qf_out = qf @ w_qe^T + b_qe + qf
  const int mtq = (FINAL_K + 15) / 16;  // 19
  k_gemm_wmma<<<dim3(cdiv(mtq, 4), C_NECK / 32), B128, gemm_shmem(C_NECK), stream>>>(
      qf_bf, FINAL_K, C_NECK, C_NECK, wqe_bf, C_NECK, b_qe, qf_f, C_NECK,
      out_qf, C_NECK, C_NECK, 0, 0);
  // qf_out += pos2embed(query_pred) @ w_qpe^T + b_qpe
  k_gemm_wmma<<<dim3(cdiv(mtq, 4), C_NECK / 32), B128, gemm_shmem(224), stream>>>(
      pe_bf, FINAL_K, 224, 224, wqpe_bf, 224, b_qpe, nullptr, 0,
      out_qf, C_NECK, C_NECK, 1, 0);
}
